// Conditional_Embedding_Contrastive_loss_32547262169719
// MI455X (gfx1250) — compile-verified, matrix operand order checked by reading, not run on hardware
//
#include <hip/hip_runtime.h>
#include <hip/hip_bf16.h>
#include <math.h>

#define NN 4096
#define DD 1024
#define CC 1000

// LDS tile row pitch: 128B of data + 16B pad = 144B (72 halves).
// Matches TDM pad config: pad_interval code 4 (32 DWORDs = 128B),
// pad_amount code 3 (4 DWORDs = 16B). 16 rows * 36 words -> 16 distinct banks.
#define PITCHH 72

typedef _Float16 v8h  __attribute__((ext_vector_type(8)));
typedef _Float16 v16h __attribute__((ext_vector_type(16)));
typedef float    v8f  __attribute__((ext_vector_type(8)));
typedef unsigned int u32;
typedef u32 v4u __attribute__((ext_vector_type(4)));
typedef int  v8i __attribute__((ext_vector_type(8)));

// ---------------------------------------------------------------------------
// Kernel 1: row norms, anchor cosine, normalized f16 copy, accumulator zeroing
// ---------------------------------------------------------------------------
__global__ __launch_bounds__(256) void normalize_kernel(
    const float* __restrict__ X, const float* __restrict__ A,
    const float* __restrict__ tptr,
    _Float16* __restrict__ Xh, float* __restrict__ pvec,
    float* __restrict__ sum1, float* __restrict__ sum2)
{
    const int row = blockIdx.x;
    const float* __restrict__ x = X + (size_t)row * DD;
    const float* __restrict__ a = A + (size_t)row * DD;

    float xx = 0.f, xa = 0.f, aa = 0.f;
    for (int c = threadIdx.x; c < DD; c += 256) {
        float xv = x[c], av = a[c];
        xx += xv * xv;
        xa += xv * av;
        aa += av * av;
    }
    #pragma unroll
    for (int m = 16; m >= 1; m >>= 1) {
        xx += __shfl_xor(xx, m, 32);
        xa += __shfl_xor(xa, m, 32);
        aa += __shfl_xor(aa, m, 32);
    }
    __shared__ float sh[3][8];
    __shared__ float s_inv;
    const int wave = threadIdx.x >> 5;
    const int lane = threadIdx.x & 31;
    if (lane == 0) { sh[0][wave] = xx; sh[1][wave] = xa; sh[2][wave] = aa; }
    __syncthreads();
    if (threadIdx.x == 0) {
        float XX = 0.f, XA = 0.f, AA = 0.f;
        #pragma unroll
        for (int w = 0; w < 8; ++w) { XX += sh[0][w]; XA += sh[1][w]; AA += sh[2][w]; }
        float nx  = sqrtf(XX);
        float na  = sqrtf(AA);
        float den = fmaxf(nx * na, 1e-8f);
        float invT = 1.0f / tptr[0];
        pvec[row] = __expf((XA / den) * invT);
        sum1[row] = 0.f;
        sum2[row] = 0.f;
        s_inv = 1.0f / fmaxf(nx, 1e-20f);
    }
    __syncthreads();
    const float invnx = s_inv;
    for (int c = threadIdx.x; c < DD; c += 256)
        Xh[(size_t)row * DD + c] = (_Float16)(x[c] * invnx);
}

// ---------------------------------------------------------------------------
// TDM: issue one tensor_load_to_lds for a 128(row) x 64(halves) tile of Xh
// into LDS at byte offset lds_off, with 16B pad per 128B row (144B pitch).
// D# layout per CDNA5 ISA ch.8 (group0 128b + group1 256b, 2-group 2D form).
// ---------------------------------------------------------------------------
__device__ __forceinline__ void tdm_load_tile(u32 lds_off, const _Float16* gptr)
{
    unsigned long long ga = (unsigned long long)(uintptr_t)gptr;
    v4u g0;
    g0.x = 1u;                                   // count=1 valid user descriptor
    g0.y = lds_off;                              // lds_addr[31:0]
    g0.z = (u32)(ga & 0xFFFFFFFFu);              // global_addr[31:0]
    g0.w = (u32)((ga >> 32) & 0x01FFFFFFu)       // global_addr[56:32]
         | (2u << 30);                           // type=2 ("image")
    v8i g1;
    g1[0] = (1 << 16)                            // data_size = 2 bytes
          | (1 << 20)                            // pad_enable
          | (4 << 22)                            // pad_interval: 32 DWORDs (128B)
          | (3 << 25);                           // pad_amount: 4 DWORDs (16B)
    g1[1] = (int)(((u32)DD & 0xFFFFu) << 16);    // tensor_dim0[15:0]
    g1[2] = (int)(((u32)DD >> 16) | (((u32)NN & 0xFFFFu) << 16)); // dim0 hi | dim1 lo
    g1[3] = (int)(((u32)NN >> 16) | (64u << 16));                 // dim1 hi | tile_dim0=64
    g1[4] = (int)128u;                           // tile_dim1=128, tile_dim2=0
    g1[5] = (int)DD;                             // tensor_dim0_stride[31:0] = 1024
    g1[6] = 0;                                   // stride hi | dim1_stride lo
    g1[7] = 0;
    asm volatile("tensor_load_to_lds %0, %1" :: "s"(g0), "s"(g1) : "memory");
}

// ---------------------------------------------------------------------------
// Fragment fetch from padded LDS tile per CDNA5 16-bit A/B layout:
//   lanes 0-15 : halves 0-7 -> k=0..7,  halves 8-15 -> k=16..23  (M/N = lane)
//   lanes16-31 : halves 0-7 -> k=8..15, halves 8-15 -> k=24..31  (M/N = lane-16)
// ---------------------------------------------------------------------------
__device__ __forceinline__ v16h lds_frag(const _Float16* __restrict__ buf,
                                         int row0, int kk, int lane)
{
    const _Float16* p = buf + (row0 + (lane & 15)) * PITCHH + kk
                            + ((lane & 16) ? 8 : 0);
    v8h lo = *(const v8h*)(p);        // ds_load_b128
    v8h hi = *(const v8h*)(p + 16);   // ds_load_b128
    v16h r;
    #pragma unroll
    for (int t = 0; t < 8; ++t) { r[t] = lo[t]; r[t + 8] = hi[t]; }
    return r;
}

// ---------------------------------------------------------------------------
// Kernel 2: Gram-matrix WMMA GEMM, TDM double-buffered LDS staging,
// fused contrastive epilogue. Block 128x128, 8 waves, wave tile 32x64.
// ---------------------------------------------------------------------------
__global__ __launch_bounds__(256) void gemm_contrast_kernel(
    const _Float16* __restrict__ Xh,
    const int* __restrict__ cls_mask, const int* __restrict__ labels,
    const float* __restrict__ tptr,
    float* __restrict__ sum1, float* __restrict__ sum2)
{
    __shared__ __align__(16) _Float16 sA[2][128 * PITCHH];  // 2 x 18KB
    __shared__ __align__(16) _Float16 sB[2][128 * PITCHH];  // 2 x 18KB

    const int lane  = threadIdx.x & 31;
    const int wave  = threadIdx.x >> 5;   // 0..7
    const int waveM = wave >> 1;          // 0..3
    const int waveN = wave & 1;           // 0..1
    const int bRow  = blockIdx.x * 128;
    const int bCol  = blockIdx.y * 128;
    const int rowBase = bRow + waveM * 32;
    const int colBase = bCol + waveN * 64;

    const float invT = 1.0f / tptr[0];

    v8f acc[2][4];
    #pragma unroll
    for (int mi = 0; mi < 2; ++mi)
        #pragma unroll
        for (int ni = 0; ni < 4; ++ni)
            acc[mi][ni] = (v8f)(0.0f);

    const int NSLAB = DD / 64;            // 16 K-slabs of 64
    // Prologue: DMA slab 0 into buffer 0 (wave 0 issues; EXEC ignored by TDM)
    if (threadIdx.x < 32) {
        tdm_load_tile((u32)(uintptr_t)&sA[0][0], Xh + (size_t)bRow * DD);
        tdm_load_tile((u32)(uintptr_t)&sB[0][0], Xh + (size_t)bCol * DD);
    }

    for (int s = 0; s < NSLAB; ++s) {
        const int cur = s & 1;
        if (threadIdx.x < 32) {
            if (s + 1 < NSLAB) {
                const int k1 = (s + 1) * 64;
                tdm_load_tile((u32)(uintptr_t)&sA[cur ^ 1][0],
                              Xh + (size_t)bRow * DD + k1);
                tdm_load_tile((u32)(uintptr_t)&sB[cur ^ 1][0],
                              Xh + (size_t)bCol * DD + k1);
                // 4 tensor ops in flight; in-order completion => cnt<=2 means
                // the current buffer's two DMAs have landed in LDS.
                __builtin_amdgcn_s_wait_tensorcnt(2);
            } else {
                __builtin_amdgcn_s_wait_tensorcnt(0);
            }
        }
        __syncthreads();                  // buffer `cur` visible to all waves

        const _Float16* __restrict__ A = sA[cur];
        const _Float16* __restrict__ B = sB[cur];
        #pragma unroll
        for (int kk = 0; kk < 64; kk += 32) {
            v16h a0 = lds_frag(A, waveM * 32,      kk, lane);
            v16h a1 = lds_frag(A, waveM * 32 + 16, kk, lane);
            v16h b0 = lds_frag(B, waveN * 64,      kk, lane);
            v16h b1 = lds_frag(B, waveN * 64 + 16, kk, lane);
            v16h b2 = lds_frag(B, waveN * 64 + 32, kk, lane);
            v16h b3 = lds_frag(B, waveN * 64 + 48, kk, lane);

            acc[0][0] = __builtin_amdgcn_wmma_f32_16x16x32_f16(false, a0, false, b0, (short)0, acc[0][0], false, false);
            acc[0][1] = __builtin_amdgcn_wmma_f32_16x16x32_f16(false, a0, false, b1, (short)0, acc[0][1], false, false);
            acc[0][2] = __builtin_amdgcn_wmma_f32_16x16x32_f16(false, a0, false, b2, (short)0, acc[0][2], false, false);
            acc[0][3] = __builtin_amdgcn_wmma_f32_16x16x32_f16(false, a0, false, b3, (short)0, acc[0][3], false, false);
            acc[1][0] = __builtin_amdgcn_wmma_f32_16x16x32_f16(false, a1, false, b0, (short)0, acc[1][0], false, false);
            acc[1][1] = __builtin_amdgcn_wmma_f32_16x16x32_f16(false, a1, false, b1, (short)0, acc[1][1], false, false);
            acc[1][2] = __builtin_amdgcn_wmma_f32_16x16x32_f16(false, a1, false, b2, (short)0, acc[1][2], false, false);
            acc[1][3] = __builtin_amdgcn_wmma_f32_16x16x32_f16(false, a1, false, b3, (short)0, acc[1][3], false, false);
        }
        __syncthreads();                  // done reading before `cur` is re-DMAed
    }

    // Epilogue. C/D layout: VGPR r, lanes 0-15 -> M=r, lanes 16-31 -> M=r+8,
    // N = lane & 15 within each 16x16 tile.
    const int halfSel = lane >> 4;
    const int nLocal  = lane & 15;

    #pragma unroll
    for (int mi = 0; mi < 2; ++mi) {
        #pragma unroll
        for (int r = 0; r < 8; ++r) {
            const int i = rowBase + mi * 16 + r + 8 * halfSel;
            const int* __restrict__ mrow = cls_mask + (size_t)labels[i] * NN;
            float s1 = 0.f, s2 = 0.f;
            #pragma unroll
            for (int ni = 0; ni < 4; ++ni) {
                const int j = colBase + ni * 16 + nLocal;
                float e = __expf(acc[mi][ni][r] * invT);
                if (i == j) e = 0.f;            // remove_diag: skip j == i
                s1 += e;
                s2 += e * (float)mrow[j];
            }
            #pragma unroll
            for (int m = 8; m >= 1; m >>= 1) {
                s1 += __shfl_xor(s1, m, 16);
                s2 += __shfl_xor(s2, m, 16);
            }
            if (nLocal == 0) {
                atomicAdd(&sum1[i], s1);
                atomicAdd(&sum2[i], s2);
            }
        }
    }
}

// ---------------------------------------------------------------------------
// Kernel 3: final log/mean reduction
// ---------------------------------------------------------------------------
__global__ __launch_bounds__(256) void finalize_kernel(
    const float* __restrict__ sum1, const float* __restrict__ sum2,
    const float* __restrict__ pvec, float* __restrict__ out)
{
    float acc = 0.f;
    for (int i = threadIdx.x; i < NN; i += 256) {
        float p = pvec[i];
        acc += __logf((sum2[i] + p) / (p + sum1[i]));
    }
    #pragma unroll
    for (int m = 16; m >= 1; m >>= 1) acc += __shfl_xor(acc, m, 32);
    __shared__ float sh[8];
    const int wave = threadIdx.x >> 5;
    const int lane = threadIdx.x & 31;
    if (lane == 0) sh[wave] = acc;
    __syncthreads();
    if (threadIdx.x == 0) {
        float t = 0.f;
        #pragma unroll
        for (int w = 0; w < 8; ++w) t += sh[w];
        out[0] = -t / (float)NN;
    }
}

// ---------------------------------------------------------------------------
extern "C" void kernel_launch(void* const* d_in, const int* in_sizes, int n_in,
                              void* d_out, int out_size, void* d_ws, size_t ws_size,
                              hipStream_t stream)
{
    const float* X        = (const float*)d_in[0];   // inst_embed [N,D] f32
    const float* A        = (const float*)d_in[1];   // anchor     [N,D] f32
    const int*   cls_mask = (const int*)  d_in[2];   // [C,N] i32
    const int*   labels   = (const int*)  d_in[3];   // [N] int
    const float* tptr     = (const float*)d_in[4];   // temperature scalar
    float* out = (float*)d_out;

    char* ws = (char*)d_ws;
    _Float16* Xh = (_Float16*)ws;                       // N*D*2 = 8 MB (normalized f16)
    float* sum1  = (float*)(ws + (size_t)NN * DD * 2);  // N floats
    float* sum2  = sum1 + NN;                           // N floats
    float* pvec  = sum2 + NN;                           // N floats

    normalize_kernel<<<NN, 256, 0, stream>>>(X, A, tptr, Xh, pvec, sum1, sum2);

    dim3 grid(NN / 128, NN / 128);
    gemm_contrast_kernel<<<grid, 256, 0, stream>>>(Xh, cls_mask, labels, tptr, sum1, sum2);

    finalize_kernel<<<1, 256, 0, stream>>>(sum1, sum2, pvec, out);
}
